// SelfAttnHead_90709709291956
// MI455X (gfx1250) — matrix operand order repaired and checked by reading, not compiled, not run on hardware
//
#include <hip/hip_runtime.h>
#include <hip/hip_bf16.h>

typedef __attribute__((ext_vector_type(16))) __bf16       v16bf;
typedef __attribute__((ext_vector_type(8)))  float        v8f;
typedef __attribute__((ext_vector_type(4)))  unsigned int v4u;
typedef __attribute__((ext_vector_type(8)))  int          v8i;
typedef __attribute__((ext_vector_type(4)))  int          v4i;

#define WMMA_BF16(a, b, c) \
  __builtin_amdgcn_wmma_f32_16x16x32_bf16(false, (a), false, (b), (short)0, (c), false, false)

static constexpr int   S    = 4096;
static constexpr int   E    = 512;
static constexpr int   H    = 64;
static constexpr float NEGF = -1e30f;

// --------------------------------------------------------------------------
// fragment load helpers (force b128 vector loads)
// --------------------------------------------------------------------------
__device__ __forceinline__ v16bf ld_frag16(const __bf16* p) {   // p[0..15]
  union { uint4 q[2]; v16bf f; } u;
  u.q[0] = *(const uint4*)p;
  u.q[1] = *(const uint4*)(p + 8);
  return u.f;
}
__device__ __forceinline__ v16bf ld_frag_a(const __bf16* p) {   // p[0..7], p[16..23]
  union { uint4 q[2]; v16bf f; } u;
  u.q[0] = *(const uint4*)p;
  u.q[1] = *(const uint4*)(p + 16);
  return u.f;
}

// --------------------------------------------------------------------------
// TDM: 2D tile load global -> LDS.  tile == tensor (no OOB), data_size = 2B.
// Descriptor per CDNA5 ISA sec.8.  Tracked by TENSORcnt.
// --------------------------------------------------------------------------
__device__ __forceinline__ void tdm_load_2d(unsigned lds_off, const void* gptr,
                                            int dim0, int dim1, int stride0) {
  unsigned long long ga = (unsigned long long)(uintptr_t)gptr;
  v4u g0;
  g0.x = 1u;                                            // count = 1 (valid)
  g0.y = lds_off;                                       // lds_addr
  g0.z = (unsigned)(ga & 0xffffffffu);                  // global_addr[31:0]
  g0.w = (unsigned)((ga >> 32) & 0x01ffffffu)           // global_addr[56:32]
         | (2u << 30);                                  // type = 2 ("image")
  v8i g1;
  g1[0] = 0x10000;                                      // data_size = 1 -> 2B
  g1[1] = (dim0 & 0xffff) << 16;                        // tensor_dim0[15:0]
  g1[2] = ((dim0 >> 16) & 0xffff) | ((dim1 & 0xffff) << 16);  // dim0 hi | dim1 lo
  g1[3] = ((dim1 >> 16) & 0xffff) | ((dim0 & 0xffff) << 16);  // dim1 hi | tile_dim0
  g1[4] = dim1 & 0xffff;                                // tile_dim1 (tile_dim2 = 0)
  g1[5] = stride0;                                      // tensor_dim0_stride[31:0]
  g1[6] = 0;
  g1[7] = 0;
  v4i gz = (v4i){0, 0, 0, 0};
#if __clang_major__ >= 23
  v8i gz8 = (v8i){0, 0, 0, 0, 0, 0, 0, 0};
  __builtin_amdgcn_tensor_load_to_lds(g0, g1, gz, gz, gz8, 0);
#else
  __builtin_amdgcn_tensor_load_to_lds(g0, g1, gz, gz, 0);
#endif
}

// --------------------------------------------------------------------------
// Kernel 1: prefix-OR of padding mask. pa[r] = any(mask[0..r] != 0).
// --------------------------------------------------------------------------
__global__ void prefix_kernel(const int* __restrict__ mask, int* __restrict__ pa) {
  const int lane = threadIdx.x & 31;
  const int base = lane * (S / 32);
  int orv = 0;
  for (int i = 0; i < S / 32; ++i) orv |= mask[base + i];
  int inc = orv;
#pragma unroll
  for (int d = 1; d < 32; d <<= 1) {
    int t = __shfl_up(inc, d, 32);
    if (lane >= d) inc |= t;
  }
  int run = __shfl_up(inc, 1, 32);
  if (lane == 0) run = 0;
  for (int i = 0; i < S / 32; ++i) {
    run |= mask[base + i];
    pa[base + i] = (run != 0) ? 1 : 0;
  }
}

// --------------------------------------------------------------------------
// Kernel 2: pack W (f32 [E][H]) -> bf16 in exact B-fragment order.
// --------------------------------------------------------------------------
__global__ __launch_bounds__(256) void packw_kernel(const float* __restrict__ W,
                                                    __bf16* __restrict__ out) {
  const int idx = blockIdx.x * 256 + threadIdx.x;       // E*H = 32768 total
  if (idx >= E * H) return;
  const int i  = idx & 15;
  const int L  = (idx >> 4) & 31;
  const int fr = idx >> 9;                              // 0..63
  const int ec = fr >> 2, t = fr & 3;
  const int hi = L >> 4, ln = L & 15;
  out[idx] = (__bf16)W[(size_t)(32 * ec + 8 * hi + (i & 7) + 16 * (i >> 3)) * H + 16 * t + ln];
}

// --------------------------------------------------------------------------
// Kernel 3: projections. q row-major bf16; v stored TRANSPOSED: vbT[b][h][s].
// --------------------------------------------------------------------------
__device__ __forceinline__ v16bf ld_a_x(const float* xp) {
  float4 f0 = *(const float4*)(xp);
  float4 f1 = *(const float4*)(xp + 4);
  float4 f2 = *(const float4*)(xp + 16);
  float4 f3 = *(const float4*)(xp + 20);
  v16bf a;
  a[0]  = (__bf16)f0.x; a[1]  = (__bf16)f0.y; a[2]  = (__bf16)f0.z; a[3]  = (__bf16)f0.w;
  a[4]  = (__bf16)f1.x; a[5]  = (__bf16)f1.y; a[6]  = (__bf16)f1.z; a[7]  = (__bf16)f1.w;
  a[8]  = (__bf16)f2.x; a[9]  = (__bf16)f2.y; a[10] = (__bf16)f2.z; a[11] = (__bf16)f2.w;
  a[12] = (__bf16)f3.x; a[13] = (__bf16)f3.y; a[14] = (__bf16)f3.z; a[15] = (__bf16)f3.w;
  return a;
}

__global__ __launch_bounds__(128) void proj_kernel(
    const float* __restrict__ x,
    const __bf16* __restrict__ wq_pk, const float* __restrict__ bq,
    const __bf16* __restrict__ wv_pk, const float* __restrict__ bv,
    __bf16* __restrict__ qb, __bf16* __restrict__ vbT) {
  const int lane = threadIdx.x & 31;
  const int wid  = threadIdx.x >> 5;
  const int tile = blockIdx.x * 4 + wid;                // 1024 tiles of 16 rows
  const int g0   = tile * 16;
  const int hi   = lane >> 4;
  const int ln   = lane & 15;

  v8f qa[4], va[4];
#pragma unroll
  for (int t = 0; t < 4; ++t) {
    qa[t] = (v8f){0.f, 0.f, 0.f, 0.f, 0.f, 0.f, 0.f, 0.f};
    va[t] = (v8f){0.f, 0.f, 0.f, 0.f, 0.f, 0.f, 0.f, 0.f};
  }

  const float* xrow = x + (size_t)(g0 + ln) * E + 8 * hi;
  v16bf a = ld_a_x(xrow);

  for (int ec = 0; ec < E / 32; ++ec) {
    v16bf bqf[4], bvf[4];
#pragma unroll
    for (int t = 0; t < 4; ++t) {
      const size_t fo = (size_t)((ec * 4 + t) * 32 + lane) * 16;
      bqf[t] = ld_frag16(wq_pk + fo);
      bvf[t] = ld_frag16(wv_pk + fo);
    }
    v16bf anext;
    if (ec + 1 < E / 32) anext = ld_a_x(xrow + 32 * (ec + 1));
#pragma unroll
    for (int t = 0; t < 4; ++t) {
      qa[t] = WMMA_BF16(a, bqf[t], qa[t]);
      va[t] = WMMA_BF16(a, bvf[t], va[t]);
    }
    a = anext;
  }

  const int batch = g0 >> 12;                           // / S
  const int s_in  = g0 & (S - 1);
#pragma unroll
  for (int t = 0; t < 4; ++t) {
    const float biasq = bq[16 * t + ln];
    const float biasv = bv[16 * t + ln];
#pragma unroll
    for (int v = 0; v < 8; ++v) {
      const int m = v + 8 * hi;
      qb[(size_t)(g0 + m) * H + 16 * t + ln] = (__bf16)(qa[t][v] + biasq);
      vbT[((size_t)batch * H + 16 * t + ln) * S + s_in + m] = (__bf16)(va[t][v] + biasv);
    }
  }
}

// --------------------------------------------------------------------------
// Kernel 4: flash attention, TDM double-buffered.  2 waves/block.
// --------------------------------------------------------------------------
__global__ __launch_bounds__(64) void flash_kernel(
    const __bf16* __restrict__ qb, const __bf16* __restrict__ vbT,
    const int* __restrict__ mask, const int* __restrict__ pa,
    float* __restrict__ out) {
  __shared__ __align__(16) __bf16 sk[2][2][32 * 64];    // [wave][buf] K [s][h]
  __shared__ __align__(16) __bf16 sv[2][2][64 * 32];    // [wave][buf] V [h][s]
  __shared__ __align__(16) __bf16 sp[2][16 * 32];       // P stage

  const int lane = threadIdx.x & 31;
  const int wid  = threadIdx.x >> 5;
  const int tile = blockIdx.x * 2 + wid;                // 1024 = B * (S/16)
  const int b    = tile >> 8;
  const int r0   = (tile & 255) << 4;
  const int hi   = lane >> 4;
  const int ln   = lane & 15;

  __bf16* spw = sp[wid];
  const unsigned lds_k0 = (unsigned)(uintptr_t)&sk[wid][0][0];
  const unsigned lds_k1 = (unsigned)(uintptr_t)&sk[wid][1][0];
  const unsigned lds_v0 = (unsigned)(uintptr_t)&sv[wid][0][0];
  const unsigned lds_v1 = (unsigned)(uintptr_t)&sv[wid][1][0];

  const __bf16* gK = qb + (size_t)b * S * H;            // K == Q (source bug)
  const __bf16* gV = vbT + (size_t)b * H * S;

  v16bf qf[2];
  {
    const __bf16* qrow = qb + (size_t)(b * S + r0 + ln) * H + 8 * hi;
    qf[0] = ld_frag_a(qrow);
    qf[1] = ld_frag_a(qrow + 32);
  }

  float mrow[8], lrow[8];
  int   dead[8];
  v8f   acc[4];
#pragma unroll
  for (int v = 0; v < 8; ++v) {
    mrow[v] = NEGF;
    lrow[v] = 0.f;
    dead[v] = (pa[r0 + v + 8 * hi] == 0);
  }
#pragma unroll
  for (int t = 0; t < 4; ++t)
    acc[t] = (v8f){0.f, 0.f, 0.f, 0.f, 0.f, 0.f, 0.f, 0.f};

  const int nkt = (r0 + 47) >> 5;                       // ceil((r0+16)/32)

  // prologue: start tile 0 into buffer 0
  tdm_load_2d(lds_k0, gK, /*dim0=*/H, /*dim1=*/32, /*stride0=*/H);
  tdm_load_2d(lds_v0, gV, /*dim0=*/32, /*dim1=*/H, /*stride0=*/S);

  for (int kt = 0; kt < nkt; ++kt) {
    const int cur = kt & 1;
    const __bf16* skw = sk[wid][cur];
    const __bf16* svw = sv[wid][cur];

    if (kt + 1 < nkt) {
      asm volatile("s_wait_dscnt 0x0" ::: "memory");    // other buf's reads done
      tdm_load_2d(cur ? lds_k0 : lds_k1, gK + (size_t)(kt + 1) * 32 * H, H, 32, H);
      tdm_load_2d(cur ? lds_v0 : lds_v1, gV + (size_t)(kt + 1) * 32,     32, H, S);
      __builtin_amdgcn_s_wait_tensorcnt(2);             // current tile landed
    } else {
      __builtin_amdgcn_s_wait_tensorcnt(0);
    }
    asm volatile("" ::: "memory");

    const int c0 = kt * 32;

    float p[2][8];
#pragma unroll
    for (int t2 = 0; t2 < 2; ++t2) {
      v8f s = (v8f){0.f, 0.f, 0.f, 0.f, 0.f, 0.f, 0.f, 0.f};
#pragma unroll
      for (int cc = 0; cc < 2; ++cc) {
        const __bf16* kr = skw + (16 * t2 + ln) * 64 + 32 * cc + 8 * hi;
        s = WMMA_BF16(qf[cc], ld_frag_a(kr), s);
      }
      const int c  = c0 + 16 * t2 + ln;
      const int mk = mask[c];
#pragma unroll
      for (int v = 0; v < 8; ++v) {
        const int r = r0 + v + 8 * hi;
        float val = ((c <= r) && mk) ? s[v] * 0.125f : NEGF;
        if (dead[v] && (c == r)) val = 1.0f;
        p[t2][v] = val;
      }
    }

#pragma unroll
    for (int v = 0; v < 8; ++v) {
      float tm = fmaxf(p[0][v], p[1][v]);
      tm = fmaxf(tm, __shfl_xor(tm, 1, 32));
      tm = fmaxf(tm, __shfl_xor(tm, 2, 32));
      tm = fmaxf(tm, __shfl_xor(tm, 4, 32));
      tm = fmaxf(tm, __shfl_xor(tm, 8, 32));
      const float mn    = fmaxf(mrow[v], tm);
      const float alpha = __expf(mrow[v] - mn);
      const float p0 = (p[0][v] > -1e29f) ? __expf(p[0][v] - mn) : 0.f;
      const float p1 = (p[1][v] > -1e29f) ? __expf(p[1][v] - mn) : 0.f;
      float rs = p0 + p1;
      rs += __shfl_xor(rs, 1, 32);
      rs += __shfl_xor(rs, 2, 32);
      rs += __shfl_xor(rs, 4, 32);
      rs += __shfl_xor(rs, 8, 32);
      lrow[v] = lrow[v] * alpha + rs;
      mrow[v] = mn;
#pragma unroll
      for (int t = 0; t < 4; ++t) acc[t][v] *= alpha;
      p[0][v] = p0;
      p[1][v] = p1;
    }

#pragma unroll
    for (int t2 = 0; t2 < 2; ++t2)
#pragma unroll
      for (int v = 0; v < 8; ++v)
        spw[(v + 8 * hi) * 32 + 16 * t2 + ln] = (__bf16)p[t2][v];

    const v16bf pf = ld_frag_a(spw + ln * 32 + 8 * hi);
#pragma unroll
    for (int t = 0; t < 4; ++t) {
      const __bf16* vr = svw + (16 * t + ln) * 32 + 8 * hi;   // V^T [h][s]
      acc[t] = WMMA_BF16(pf, ld_frag_a(vr), acc[t]);
    }
  }

#pragma unroll
  for (int v = 0; v < 8; ++v) {
    const float inv = 1.0f / lrow[v];
    const size_t rowo = (size_t)(b * S + r0 + v + 8 * hi) * H;
#pragma unroll
    for (int t = 0; t < 4; ++t)
      out[rowo + 16 * t + ln] = acc[t][v] * inv;
  }
}

// --------------------------------------------------------------------------
extern "C" void kernel_launch(void* const* d_in, const int* in_sizes, int n_in,
                              void* d_out, int out_size, void* d_ws, size_t ws_size,
                              hipStream_t stream) {
  (void)in_sizes; (void)n_in; (void)out_size; (void)ws_size;

  const float* x   = (const float*)d_in[0];
  const float* Wq  = (const float*)d_in[1];
  const float* bq  = (const float*)d_in[2];
  const float* Wv  = (const float*)d_in[3];
  const float* bv  = (const float*)d_in[4];
  const int*   msk = (const int*)d_in[5];
  float*       out = (float*)d_out;

  char* ws = (char*)d_ws;
  __bf16* qb    = (__bf16*)(ws);
  __bf16* vbT   = (__bf16*)(ws + (size_t)2 * 1024 * 1024);
  int*    pa    = (int*)(ws + (size_t)4 * 1024 * 1024);
  __bf16* wq_pk = (__bf16*)(ws + (size_t)4 * 1024 * 1024 + 16 * 1024);
  __bf16* wv_pk = (__bf16*)(ws + (size_t)4 * 1024 * 1024 + 16 * 1024 + 64 * 1024);

  prefix_kernel<<<1, 32, 0, stream>>>(msk, pa);
  packw_kernel<<<(E * H + 255) / 256, 256, 0, stream>>>(Wq, wq_pk);
  packw_kernel<<<(E * H + 255) / 256, 256, 0, stream>>>(Wv, wv_pk);
  proj_kernel<<<256, 128, 0, stream>>>(x, wq_pk, bq, wv_pk, bv, qb, vbT);
  flash_kernel<<<512, 64, 0, stream>>>(qb, vbT, msk, pa, out);
}
